// AdaptivePyramidPool_3204045603309
// MI455X (gfx1250) — matrix operand order repaired and checked by reading, not compiled, not run on hardware
//
#include <hip/hip_runtime.h>
#include <hip/hip_bf16.h>

typedef __attribute__((ext_vector_type(16))) __bf16 v16bf;
typedef __attribute__((ext_vector_type(8)))  float  v8f;

#define BATCH   32
#define SEQ     4096
#define DMODEL  768
#define AATT    128
#define NS      3
#define NCOLS   (NS * AATT)       // 384
#define KTILES  (DMODEL / 32)     // 24
#define NTILES  (NCOLS / 16)      // 24
#define ROWS    (BATCH * SEQ)     // 131072
#define ROWTILES (ROWS / 16)      // 8192

static __device__ __forceinline__ unsigned short f2bf(float f) {
    union { float f; unsigned u; } c; c.f = f;
    unsigned r = c.u + 0x7FFFu + ((c.u >> 16) & 1u);   // round-to-nearest-even
    return (unsigned short)(r >> 16);
}

// ---------------------------------------------------------------------------
// Kernel P: pack Wp [3,768,128] fp32 -> bf16 WMMA B-fragment layout.
// Dest index = ((kt*NTILES + nt)*32 + lane)*16 + i
//   col n = nt*16 + (lane&15); scale s = n>>7; a = n&127
//   K     = kt*32 + (lane>=16 ? 8 : 0) + (i&7) + (i>=8 ? 16 : 0)     (B 32x16 bf16 layout, ISA 7.12.2)
// ---------------------------------------------------------------------------
__global__ void pack_wp_kernel(const float* __restrict__ Wp,
                               unsigned short* __restrict__ pb) {
    int tid = blockIdx.x * blockDim.x + threadIdx.x;
    if (tid >= KTILES * NTILES * 32 * 16) return;
    int i    = tid & 15;
    int lane = (tid >> 4) & 31;
    int rest = tid >> 9;
    int nt   = rest % NTILES;
    int kt   = rest / NTILES;
    int n = nt * 16 + (lane & 15);
    int s = n >> 7;
    int a = n & 127;
    int k = kt * 32 + ((lane & 16) ? 8 : 0) + (i & 7) + ((i & 8) ? 16 : 0);
    pb[tid] = f2bf(Wp[((size_t)s * DMODEL + k) * AATT + a]);
}

__global__ void zero_kernel(float* __restrict__ p, int n) {
    int i = blockIdx.x * blockDim.x + threadIdx.x;
    if (i < n) p[i] = 0.0f;
}

// ---------------------------------------------------------------------------
// Kernel A: WMMA GEMM  scores[row][s] = dot(tanh(x_row @ Wp[s] + bp[s]), v[s])
// One wave per 16-row tile. x tile staged in LDS as bf16 (24 KB/wave).
// 6 N-groups (each = 64 cols = half a scale) x 24 K-steps x 4 WMMA.
// ---------------------------------------------------------------------------
__global__ __launch_bounds__(64)
void score_gemm_kernel(const float* __restrict__ x,
                       const unsigned short* __restrict__ pb,
                       const float* __restrict__ bp,
                       const float* __restrict__ v,
                       float* __restrict__ scores) {
    __shared__ unsigned short xs[2][16 * DMODEL];   // 48 KB
    const int lane = threadIdx.x & 31;
    const int w    = threadIdx.x >> 5;
    const int tile = blockIdx.x * 2 + w;
    const size_t rowbase = (size_t)tile * 16;

    // ---- stage 16x768 fp32 -> bf16 into LDS (coalesced float4 loads) ----
    {
        const float* xt = x + rowbase * DMODEL;
        unsigned short* dst = xs[w];
        for (int i = lane * 4; i < 16 * DMODEL; i += 32 * 4) {
            float4 f = *(const float4*)(xt + i);
            dst[i + 0] = f2bf(f.x);
            dst[i + 1] = f2bf(f.y);
            dst[i + 2] = f2bf(f.z);
            dst[i + 3] = f2bf(f.w);
        }
    }
    __syncthreads();

    union U { uint4 u[2]; v16bf v; };
    const int m     = lane & 15;                 // matrix row within tile
    const int khalf = (lane & 16) ? 8 : 0;       // A 16x32 lane-half K interleave
    const unsigned short* arow = &xs[w][m * DMODEL];

    float scrp[8];
    #pragma unroll
    for (int r = 0; r < 8; ++r) scrp[r] = 0.0f;

    for (int ng = 0; ng < 6; ++ng) {             // 6 groups of 4 N-tiles
        v8f acc[4];
        #pragma unroll
        for (int ni = 0; ni < 4; ++ni) acc[ni] = (v8f){0,0,0,0,0,0,0,0};

        for (int k = 0; k < KTILES; ++k) {
            U A;
            const unsigned short* ap = arow + k * 32 + khalf;
            A.u[0] = *(const uint4*)(ap);        // K: kbase+{0..7} or +{8..15}
            A.u[1] = *(const uint4*)(ap + 16);   // K: +{16..23} or +{24..31}
            const uint4* q = (const uint4*)pb +
                ((size_t)(k * NTILES + ng * 4) * 32 + lane) * 2;
            #pragma unroll
            for (int ni = 0; ni < 4; ++ni) {
                U B;
                B.u[0] = q[ni * 64];             // consecutive nt: +32 lanes*32B
                B.u[1] = q[ni * 64 + 1];
                acc[ni] = __builtin_amdgcn_wmma_f32_16x16x32_bf16(
                    false, A.v, false, B.v, (short)0, acc[ni], false, false);
            }
        }

        // ---- epilogue: bias, tanh, dot with v (per-lane N column) ----
        const int s = ng >> 1;                   // group -> scale
        #pragma unroll
        for (int ni = 0; ni < 4; ++ni) {
            int col = (ng * 4 + ni) * 16 + m;    // C/D: N = lane%16 within tile
            int a   = col & 127;
            float vv  = v[s * AATT + a];
            float bpv = bp[s * AATT + a];
            #pragma unroll
            for (int r = 0; r < 8; ++r)
                scrp[r] += tanhf(acc[ni][r] + bpv) * vv;
        }

        if (ng & 1) {                            // scale complete -> reduce/store
            #pragma unroll
            for (int r = 0; r < 8; ++r) {
                float t = scrp[r];
                t += __shfl_xor(t, 1, 16);
                t += __shfl_xor(t, 2, 16);
                t += __shfl_xor(t, 4, 16);
                t += __shfl_xor(t, 8, 16);
                scrp[r] = t;
            }
            if ((lane & 15) == 0) {
                // lanes 0..15 hold M=r, lanes 16..31 hold M=8+r (C/D layout)
                size_t rb = rowbase + ((lane & 16) ? 8 : 0);
                #pragma unroll
                for (int r = 0; r < 8; ++r)
                    scores[(rb + r) * 4 + s] = scrp[r];
            }
            #pragma unroll
            for (int r = 0; r < 8; ++r) scrp[r] = 0.0f;
        }
    }
}

// ---------------------------------------------------------------------------
// Kernel B: window softmax + attention pooling + mean over windows.
// Block = (b, 128-row chunk); 256 threads cover D=768 (3 cols each).
// ---------------------------------------------------------------------------
__global__ __launch_bounds__(256)
void pool_accum_kernel(const float* __restrict__ x,
                       const float* __restrict__ scores,
                       float* __restrict__ feats) {
    __shared__ float coeff[128][3];
    const int b   = blockIdx.y;
    const int t0  = blockIdx.x * 128;
    const int tid = threadIdx.x;

    if (tid < 16) {                              // each handles one 8-row group
        const int tg = t0 + tid * 8;
        float sc[8][3];
        for (int j = 0; j < 8; ++j)
            for (int s = 0; s < 3; ++s)
                sc[j][s] = scores[((size_t)b * SEQ + tg + j) * 4 + s];
        const int ps[3] = {2, 4, 8};
        for (int s = 0; s < 3; ++s) {
            int p = ps[s];
            for (int w0 = 0; w0 < 8; w0 += p) {
                float mx = -1e30f;
                for (int j = 0; j < p; ++j) mx = fmaxf(mx, sc[w0 + j][s]);
                float e[8]; float sum = 0.0f;
                for (int j = 0; j < p; ++j) { e[j] = __expf(sc[w0 + j][s] - mx); sum += e[j]; }
                float inv = 1.0f / sum;
                for (int j = 0; j < p; ++j) coeff[tid * 8 + w0 + j][s] = e[j] * inv;
            }
        }
    }
    __syncthreads();

    float a00=0,a01=0,a02=0, a10=0,a11=0,a12=0, a20=0,a21=0,a22=0;
    const float* xb = x + ((size_t)b * SEQ + t0) * DMODEL;
    for (int j = 0; j < 128; ++j) {
        const float* xr = xb + (size_t)j * DMODEL;
        float c0 = coeff[j][0], c1 = coeff[j][1], c2 = coeff[j][2];
        float x0 = xr[tid], x1 = xr[tid + 256], x2 = xr[tid + 512];
        a00 += c0 * x0; a01 += c0 * x1; a02 += c0 * x2;
        a10 += c1 * x0; a11 += c1 * x1; a12 += c1 * x2;
        a20 += c2 * x0; a21 += c2 * x1; a22 += c2 * x2;
    }
    const float iw0 = 1.0f / 2048.0f, iw1 = 1.0f / 1024.0f, iw2 = 1.0f / 512.0f;
    float* fb = feats + (size_t)b * (3 * DMODEL);
    atomicAdd(fb + 0 * DMODEL + tid,       a00 * iw0);
    atomicAdd(fb + 0 * DMODEL + tid + 256, a01 * iw0);
    atomicAdd(fb + 0 * DMODEL + tid + 512, a02 * iw0);
    atomicAdd(fb + 1 * DMODEL + tid,       a10 * iw1);
    atomicAdd(fb + 1 * DMODEL + tid + 256, a11 * iw1);
    atomicAdd(fb + 1 * DMODEL + tid + 512, a12 * iw1);
    atomicAdd(fb + 2 * DMODEL + tid,       a20 * iw2);
    atomicAdd(fb + 2 * DMODEL + tid + 256, a21 * iw2);
    atomicAdd(fb + 2 * DMODEL + tid + 512, a22 * iw2);
}

// ---------------------------------------------------------------------------
// Kernel C: ms = feats @ Wf + bf ; LayerNorm -> out. One block per batch row.
// ---------------------------------------------------------------------------
__global__ __launch_bounds__(256)
void final_ln_kernel(const float* __restrict__ feats,
                     const float* __restrict__ Wf,
                     const float* __restrict__ bfv,
                     const float* __restrict__ gamma,
                     const float* __restrict__ beta,
                     float* __restrict__ out) {
    __shared__ float sh[3 * DMODEL];
    __shared__ float ms[DMODEL];
    __shared__ float red[256], red2[256];
    const int b = blockIdx.x, tid = threadIdx.x;

    for (int i = tid; i < 3 * DMODEL; i += 256) sh[i] = feats[(size_t)b * (3 * DMODEL) + i];
    __syncthreads();

    #pragma unroll
    for (int slot = 0; slot < 3; ++slot) {
        int d = tid + slot * 256;
        float acc = bfv[d];
        for (int k = 0; k < 3 * DMODEL; ++k) acc += sh[k] * Wf[(size_t)k * DMODEL + d];
        ms[d] = acc;
    }
    __syncthreads();

    float p = ms[tid] + ms[tid + 256] + ms[tid + 512];
    float q = ms[tid] * ms[tid] + ms[tid + 256] * ms[tid + 256] + ms[tid + 512] * ms[tid + 512];
    red[tid] = p; red2[tid] = q;
    __syncthreads();
    for (int off = 128; off > 0; off >>= 1) {
        if (tid < off) { red[tid] += red[tid + off]; red2[tid] += red2[tid + off]; }
        __syncthreads();
    }
    float mu  = red[0] * (1.0f / DMODEL);
    float var = red2[0] * (1.0f / DMODEL) - mu * mu;
    float rs  = rsqrtf(var + 1e-5f);
    #pragma unroll
    for (int slot = 0; slot < 3; ++slot) {
        int d = tid + slot * 256;
        out[(size_t)b * DMODEL + d] = (ms[d] - mu) * rs * gamma[d] + beta[d];
    }
}

// ---------------------------------------------------------------------------
extern "C" void kernel_launch(void* const* d_in, const int* in_sizes, int n_in,
                              void* d_out, int out_size, void* d_ws, size_t ws_size,
                              hipStream_t stream) {
    const float* x     = (const float*)d_in[0];
    const float* Wp    = (const float*)d_in[1];
    const float* bp    = (const float*)d_in[2];
    const float* v     = (const float*)d_in[3];
    const float* Wf    = (const float*)d_in[4];
    const float* bfv   = (const float*)d_in[5];
    const float* gamma = (const float*)d_in[6];
    const float* beta  = (const float*)d_in[7];
    float* out = (float*)d_out;

    char* ws = (char*)d_ws;
    unsigned short* pb = (unsigned short*)ws;                    // 589,824 B
    float* scores = (float*)(ws + (1u << 20));                   // 2 MiB
    float* feats  = (float*)(ws + (1u << 20) + (2u << 20));      // 294,912 B

    // 1) pack Wp into WMMA B-fragment layout
    {
        int n = KTILES * NTILES * 32 * 16;
        pack_wp_kernel<<<(n + 255) / 256, 256, 0, stream>>>(Wp, pb);
    }
    // 2) zero feats accumulator
    {
        int n = BATCH * 3 * DMODEL;
        zero_kernel<<<(n + 255) / 256, 256, 0, stream>>>(feats, n);
    }
    // 3) WMMA GEMM + tanh + score dot
    score_gemm_kernel<<<ROWTILES / 2, 64, 0, stream>>>(x, pb, bp, v, scores);
    // 4) window softmax + attention pooling (mean over windows via atomics)
    {
        dim3 grid(SEQ / 128, BATCH);
        pool_accum_kernel<<<grid, 256, 0, stream>>>(x, scores, feats);
    }
    // 5) final projection + LayerNorm
    final_ln_kernel<<<BATCH, 256, 0, stream>>>(feats, Wf, bfv, gamma, beta, out);
}
